// Model_9500467659336
// MI455X (gfx1250) — compile-verified, matrix operand order checked by reading, not run on hardware
//
#include <hip/hip_runtime.h>

#define DMODEL 256
#define NHEAD  8
#define DHEAD  32
#define FFDIM  2048
#define NLAYER 6
#define BATCH  4
#define SEQ    1022
#define NTOK   1024
#define MROWS  (BATCH * NTOK)   // 4096

typedef __attribute__((ext_vector_type(16))) __bf16 v16bf;
typedef __attribute__((ext_vector_type(8)))  __bf16 v8bf_t;
typedef __attribute__((ext_vector_type(8)))  float  v8f;

__device__ __forceinline__ unsigned short f2bfu(float f) {
  unsigned u = __float_as_uint(f);
  u += 0x7FFFu + ((u >> 16) & 1u);   // round-to-nearest-even
  return (unsigned short)(u >> 16);
}

// Load a 16x32 bf16 A-fragment (or 32x16 B-fragment when the operand is a
// row-major [16 x K] tile) from row-major memory.
// CDNA5 layout: lane<16 holds row (lane&15), K = 0..7 and 16..23;
//               lane>=16 holds same row,     K = 8..15 and 24..31.
__device__ __forceinline__ v16bf load_frag_b16(const unsigned short* base, int ld) {
  int lane = threadIdx.x & 31;
  const unsigned short* p = base + (size_t)(lane & 15) * ld + ((lane >> 4) << 3);
  v8bf_t lo = *reinterpret_cast<const v8bf_t*>(p);
  v8bf_t hi = *reinterpret_cast<const v8bf_t*>(p + 16);
  v16bf out;
#pragma unroll
  for (int i = 0; i < 8; ++i) { out[i] = lo[i]; out[8 + i] = hi[i]; }
  return out;
}

__device__ __forceinline__ v8f wmma_bf16(v16bf a, v16bf b, v8f c) {
  return __builtin_amdgcn_wmma_f32_16x16x32_bf16(false, a, false, b, (short)0, c,
                                                 false, false);
}

// Async global -> LDS copy of 16 bytes (GVS mode: SGPR base + 32-bit VGPR
// byte offset). Tracked by ASYNCcnt.
__device__ __forceinline__ void async_g2l_b128(unsigned lds_addr, unsigned gofs,
                                               const unsigned short* sbase) {
  asm volatile("global_load_async_to_lds_b128 %0, %1, %2"
               :: "v"(lds_addr), "v"(gofs), "s"(sbase) : "memory");
}
__device__ __forceinline__ void wait_async0() {
  asm volatile("s_wait_asynccnt 0x0" ::: "memory");
}

// ---------------------------------------------------------------- weights cvt
__global__ void cvt_kernel(const float* __restrict__ src,
                           unsigned short* __restrict__ dst, int n) {
  int i = blockIdx.x * 256 + threadIdx.x;
  if (i < n) dst[i] = f2bfu(src[i]);
}

// ------------------------------------------------------------- embed + posenc
__global__ __launch_bounds__(256) void embed_kernel(
    const float* __restrict__ x, const float* __restrict__ cls1,
    const float* __restrict__ cls2, const int* __restrict__ ind,
    float* __restrict__ h, unsigned short* __restrict__ hb) {
  int gid = blockIdx.x * 256 + threadIdx.x;       // covers B*N*D exactly
  int b = gid >> 18;                              // N*D = 2^18
  int p = (gid >> 8) & (NTOK - 1);
  int d = gid & (DMODEL - 1);
  int idx = ind[b];
  float tok;
  if (p == 0)            tok = cls1[d];
  else if (p <= idx)     tok = x[((size_t)b * SEQ + (p - 1)) * DMODEL + d];
  else if (p == idx + 1) tok = cls2[d];
  else                   tok = x[((size_t)b * SEQ + (p - 2)) * DMODEL + d];
  int i2 = d & ~1;
  float freq = __expf(-0.03597789063f * (float)i2);  // -ln(10000)/256 * i2
  float ang = (float)p * freq;
  float pe = (d & 1) ? __cosf(ang) : __sinf(ang);
  float v = tok + pe;
  h[gid] = v;
  hb[gid] = f2bfu(v);
}

// ----------------------------------------------------------------- GEMM tiles
// C = A[M,K] @ W[N,K]^T + bias.  MODE 0: fp32 out. MODE 1: relu->bf16 out.
// MODE 2: QKV scatter (Q,K as [B,H,N,DH] bf16, V transposed [B,H,DH,N] bf16).
template <int MODE>
__device__ __forceinline__ void store_frag(
    v8f c, int mbase, int nbase, const float* __restrict__ bias, int Nout,
    float* __restrict__ outf, unsigned short* __restrict__ outb,
    unsigned short* __restrict__ Qb, unsigned short* __restrict__ Kb,
    unsigned short* __restrict__ Vt) {
  int lane = threadIdx.x & 31;
  int n = nbase + (lane & 15);
  int m0 = mbase + ((lane >> 4) << 3);
  float bv = bias[n];
#pragma unroll
  for (int r = 0; r < 8; ++r) {
    float v = c[r] + bv;
    int m = m0 + r;
    if constexpr (MODE == 0) {
      outf[(size_t)m * Nout + n] = v;
    } else if constexpr (MODE == 1) {
      outb[(size_t)m * Nout + n] = f2bfu(fmaxf(v, 0.f));
    } else {
      int region = n >> 8;              // 0=Q 1=K 2=V
      int dcol = n & 255;
      int head = dcol >> 5;
      int dh = dcol & 31;
      int bb = m >> 10;
      int srow = m & (NTOK - 1);
      unsigned short val = f2bfu(v);
      size_t bh = (size_t)(bb * NHEAD + head);
      if (region == 0)      Qb[(bh * NTOK + srow) * DHEAD + dh] = val;
      else if (region == 1) Kb[(bh * NTOK + srow) * DHEAD + dh] = val;
      else                  Vt[(bh * DHEAD + dh) * NTOK + srow] = val;
    }
  }
}

// Block tile 128(M) x 64(N) x 32(K), double-buffered in LDS via async
// global->LDS copies (ASYNCcnt). 8 waves arranged 4(m) x 2(n), each computing
// a 32x32 WMMA tile. Copy of step i+1 overlaps WMMAs of step i.
template <int MODE>
__global__ __launch_bounds__(256) void gemm_kernel(
    const unsigned short* __restrict__ A, const unsigned short* __restrict__ W,
    const float* __restrict__ bias, int K, int Nout,
    float* __restrict__ outf, unsigned short* __restrict__ outb,
    unsigned short* __restrict__ Qb, unsigned short* __restrict__ Kb,
    unsigned short* __restrict__ Vt) {
  __shared__ __align__(16) unsigned short tA[2][128][40];
  __shared__ __align__(16) unsigned short tB[2][64][40];
  int t = threadIdx.x;
  int w = t >> 5;
  int mB = blockIdx.y * 128;
  int nB = blockIdx.x * 64;
  int mw = (w >> 1) * 32;   // within block tile
  int nw = (w & 1) * 32;

  // per-thread copy slots: 16B chunks; A = 512 chunks (2/thread), B = 256 (1/thread)
  int rowA = t >> 2;            // 0..63
  int ch = (t & 3) * 8;         // element offset of 16B chunk
  unsigned aoff0 = ((unsigned)(mB + rowA) * (unsigned)K + (unsigned)ch) * 2u;
  unsigned aoff1 = aoff0 + (unsigned)(64 * K) * 2u;
  unsigned boff  = ((unsigned)(nB + rowA) * (unsigned)K + (unsigned)ch) * 2u;
  unsigned ldsA0[2], ldsA1[2], ldsB[2];
#pragma unroll
  for (int bf = 0; bf < 2; ++bf) {
    ldsA0[bf] = (unsigned)(uintptr_t)&tA[bf][rowA][ch];
    ldsA1[bf] = (unsigned)(uintptr_t)&tA[bf][64 + rowA][ch];
    ldsB[bf]  = (unsigned)(uintptr_t)&tB[bf][rowA][ch];
  }

  auto stage = [&](int k0, int bf) {
    unsigned kb = (unsigned)k0 * 2u;
    async_g2l_b128(ldsA0[bf], aoff0 + kb, A);
    async_g2l_b128(ldsA1[bf], aoff1 + kb, A);
    async_g2l_b128(ldsB[bf], boff + kb, W);
  };

  v8f c00 = {}, c01 = {}, c10 = {}, c11 = {};
  int nsteps = K / 32;
  stage(0, 0);
  for (int i = 0; i < nsteps; ++i) {
    wait_async0();          // this wave's tile-i copies landed in LDS
    __syncthreads();        // whole tile visible; all waves done with tile i-1
    if (i + 1 < nsteps) stage((i + 1) * 32, (i + 1) & 1);
    int bf = i & 1;
    v16bf a0 = load_frag_b16(&tA[bf][mw][0], 40);
    v16bf a1 = load_frag_b16(&tA[bf][mw + 16][0], 40);
    v16bf b0 = load_frag_b16(&tB[bf][nw][0], 40);
    v16bf b1 = load_frag_b16(&tB[bf][nw + 16][0], 40);
    c00 = wmma_bf16(a0, b0, c00);
    c01 = wmma_bf16(a0, b1, c01);
    c10 = wmma_bf16(a1, b0, c10);
    c11 = wmma_bf16(a1, b1, c11);
  }
  store_frag<MODE>(c00, mB + mw,      nB + nw,      bias, Nout, outf, outb, Qb, Kb, Vt);
  store_frag<MODE>(c01, mB + mw,      nB + nw + 16, bias, Nout, outf, outb, Qb, Kb, Vt);
  store_frag<MODE>(c10, mB + mw + 16, nB + nw,      bias, Nout, outf, outb, Qb, Kb, Vt);
  store_frag<MODE>(c11, mB + mw + 16, nB + nw + 16, bias, Nout, outf, outb, Qb, Kb, Vt);
}

// ------------------------------------------------------- flash attention core
// Each wave: 16 query rows, online softmax over 64-key blocks (amortizes the
// cross-lane reductions and exp corrections over 8 WMMAs per block).
__global__ __launch_bounds__(256) void attention_kernel(
    const unsigned short* __restrict__ Qb, const unsigned short* __restrict__ Kb,
    const unsigned short* __restrict__ Vt, const int* __restrict__ ind,
    unsigned short* __restrict__ Ab) {
  __shared__ __align__(16) unsigned short pbuf[8][16][72];  // per-wave 16x64 P
  int w = threadIdx.x >> 5;
  int lane = threadIdx.x & 31;
  int qb = (int)blockIdx.x * 128 + w * 16;
  int hh = blockIdx.y;
  int bb = blockIdx.z;
  size_t bh = (size_t)(bb * NHEAD + hh);

  int idx = ind[bb];
  int pad = ind[BATCH + bb];
  int lim2 = NTOK - pad;

  v16bf qf = load_frag_b16(Qb + (bh * NTOK + qb) * DHEAD, DHEAD);

  v8f o0 = {}, o1 = {};
  float mrow[8], lrow[8];
  bool qv1[8], qv2[8];
#pragma unroll
  for (int r = 0; r < 8; ++r) {
    mrow[r] = -3.0e38f;
    lrow[r] = 0.f;
    int q = qb + ((lane >> 4) << 3) + r;
    qv1[r] = (q <= idx);
    qv2[r] = (q > idx) && (q < lim2);
  }
  const float scale = 0.17677669529663687f;  // 1/sqrt(DH)
  int lc = lane & 15;
  int moff = (lane >> 4) << 3;

  for (int j = 0; j < NTOK / 64; ++j) {
    int k0 = j * 64;
    v16bf kf0 = load_frag_b16(Kb + (bh * NTOK + k0) * DHEAD, DHEAD);
    v16bf kf1 = load_frag_b16(Kb + (bh * NTOK + k0 + 16) * DHEAD, DHEAD);
    v16bf kf2 = load_frag_b16(Kb + (bh * NTOK + k0 + 32) * DHEAD, DHEAD);
    v16bf kf3 = load_frag_b16(Kb + (bh * NTOK + k0 + 48) * DHEAD, DHEAD);
    v8f s0 = {}, s1 = {}, s2 = {}, s3 = {};
    s0 = wmma_bf16(qf, kf0, s0);
    s1 = wmma_bf16(qf, kf1, s1);
    s2 = wmma_bf16(qf, kf2, s2);
    s3 = wmma_bf16(qf, kf3, s3);

    // per-column key flags (row-independent part)
    bool kv1[4], kv2[4];
#pragma unroll
    for (int i = 0; i < 4; ++i) {
      int kc = k0 + i * 16 + lc;
      kv1[i] = (kc <= idx);
      kv2[i] = (kc > idx) && (kc < lim2);
    }
#pragma unroll
    for (int r = 0; r < 8; ++r) {
      float sv0 = s0[r] * scale + (((qv1[r] && kv1[0]) || (qv2[r] && kv2[0])) ? 0.f : -1e9f);
      float sv1 = s1[r] * scale + (((qv1[r] && kv1[1]) || (qv2[r] && kv2[1])) ? 0.f : -1e9f);
      float sv2 = s2[r] * scale + (((qv1[r] && kv1[2]) || (qv2[r] && kv2[2])) ? 0.f : -1e9f);
      float sv3 = s3[r] * scale + (((qv1[r] && kv1[3]) || (qv2[r] && kv2[3])) ? 0.f : -1e9f);
      float mx = fmaxf(fmaxf(sv0, sv1), fmaxf(sv2, sv3));
#pragma unroll
      for (int d = 1; d < 16; d <<= 1) mx = fmaxf(mx, __shfl_xor(mx, d, 16));
      float mnew = fmaxf(mrow[r], mx);
      float corr = __expf(mrow[r] - mnew);
      float p0 = __expf(sv0 - mnew);
      float p1 = __expf(sv1 - mnew);
      float p2 = __expf(sv2 - mnew);
      float p3 = __expf(sv3 - mnew);
      float rs = (p0 + p1) + (p2 + p3);
#pragma unroll
      for (int d = 1; d < 16; d <<= 1) rs += __shfl_xor(rs, d, 16);
      lrow[r] = lrow[r] * corr + rs;
      mrow[r] = mnew;
      o0[r] *= corr;
      o1[r] *= corr;
      int row = moff + r;
      pbuf[w][row][lc]      = f2bfu(p0);
      pbuf[w][row][lc + 16] = f2bfu(p1);
      pbuf[w][row][lc + 32] = f2bfu(p2);
      pbuf[w][row][lc + 48] = f2bfu(p3);
    }
    // wave-internal LDS layout swap: C-layout scores -> A-layout P fragments
    asm volatile("s_wait_dscnt 0x0" ::: "memory");
    v16bf pf0 = load_frag_b16(&pbuf[w][0][0], 72);
    v16bf pf1 = load_frag_b16(&pbuf[w][0][32], 72);
    asm volatile("" ::: "memory");
    v16bf v00 = load_frag_b16(Vt + (bh * DHEAD + 0) * NTOK + k0, NTOK);
    v16bf v01 = load_frag_b16(Vt + (bh * DHEAD + 0) * NTOK + k0 + 32, NTOK);
    v16bf v10 = load_frag_b16(Vt + (bh * DHEAD + 16) * NTOK + k0, NTOK);
    v16bf v11 = load_frag_b16(Vt + (bh * DHEAD + 16) * NTOK + k0 + 32, NTOK);
    o0 = wmma_bf16(pf0, v00, o0);
    o0 = wmma_bf16(pf1, v01, o0);
    o1 = wmma_bf16(pf0, v10, o1);
    o1 = wmma_bf16(pf1, v11, o1);
  }
#pragma unroll
  for (int r = 0; r < 8; ++r) {
    int q = qb + moff + r;
    float inv = 1.f / lrow[r];
    size_t base = ((size_t)(bb * NTOK + q)) * DMODEL + hh * DHEAD;
    Ab[base + lc] = f2bfu(o0[r] * inv);
    Ab[base + 16 + lc] = f2bfu(o1[r] * inv);
  }
}

// ------------------------------------------------------ residual + layer norm
__global__ __launch_bounds__(256) void add_ln_kernel(
    float* __restrict__ h, const float* __restrict__ y,
    const float* __restrict__ g, const float* __restrict__ bta,
    unsigned short* __restrict__ hb) {
  __shared__ float red[16];
  int t = threadIdx.x;
  size_t o = (size_t)blockIdx.x * DMODEL + t;
  float v = h[o] + y[o];
  float s = v;
#pragma unroll
  for (int d = 1; d < 32; d <<= 1) s += __shfl_xor(s, d, 32);
  int w = t >> 5, lane = t & 31;
  if (lane == 0) red[w] = s;
  __syncthreads();
  float tot = 0.f;
#pragma unroll
  for (int i = 0; i < 8; ++i) tot += red[i];
  float mean = tot * (1.f / DMODEL);
  float dv = v - mean;
  float s2 = dv * dv;
#pragma unroll
  for (int d = 1; d < 32; d <<= 1) s2 += __shfl_xor(s2, d, 32);
  if (lane == 0) red[8 + w] = s2;
  __syncthreads();
  float tot2 = 0.f;
#pragma unroll
  for (int i = 0; i < 8; ++i) tot2 += red[8 + i];
  float rstd = rsqrtf(tot2 * (1.f / DMODEL) + 1e-5f);
  float out = dv * rstd * g[t] + bta[t];
  h[o] = out;
  hb[o] = f2bfu(out);
}

// ----------------------------------------------------------------- cls gather
__global__ void gather_kernel(const float* __restrict__ h,
                              const int* __restrict__ ind,
                              float* __restrict__ out) {
  int t = blockIdx.x * 256 + threadIdx.x;
  if (t >= BATCH * 2 * DMODEL) return;
  int b = t >> 9;            // 2*256 per batch
  int which = (t >> 8) & 1;
  int d = t & 255;
  int pos = which ? (ind[b] + 1) : 0;
  out[t] = h[((size_t)(b * NTOK + pos)) * DMODEL + d];
}

// =============================================================== host driver
extern "C" void kernel_launch(void* const* d_in, const int* in_sizes, int n_in,
                              void* d_out, int out_size, void* d_ws,
                              size_t ws_size, hipStream_t stream) {
  (void)in_sizes; (void)n_in; (void)out_size; (void)ws_size;
  const float* x    = (const float*)d_in[0];
  const int*   ind  = (const int*)d_in[2];
  const float* cls1 = (const float*)d_in[3];
  const float* cls2 = (const float*)d_in[4];
  const float* Wqkv = (const float*)d_in[5];
  const float* bqkv = (const float*)d_in[6];
  const float* Wo   = (const float*)d_in[7];
  const float* bo   = (const float*)d_in[8];
  const float* ln1g = (const float*)d_in[9];
  const float* ln1b = (const float*)d_in[10];
  const float* W1   = (const float*)d_in[11];
  const float* b1   = (const float*)d_in[12];
  const float* W2   = (const float*)d_in[13];
  const float* b2   = (const float*)d_in[14];
  const float* ln2g = (const float*)d_in[15];
  const float* ln2b = (const float*)d_in[16];

  char* ws = (char*)d_ws;
  size_t off = 0;
  auto alloc = [&](size_t bytes) -> void* {
    off = (off + 255) & ~(size_t)255;
    void* p = ws + off;
    off += bytes;
    return p;
  };
  float* h            = (float*)alloc((size_t)MROWS * DMODEL * 4);
  unsigned short* hb  = (unsigned short*)alloc((size_t)MROWS * DMODEL * 2);
  float* ybuf         = (float*)alloc((size_t)MROWS * DMODEL * 4);
  unsigned short* ab  = (unsigned short*)alloc((size_t)MROWS * DMODEL * 2);
  unsigned short* Qb  = (unsigned short*)alloc((size_t)BATCH * NHEAD * NTOK * DHEAD * 2);
  unsigned short* Kb  = (unsigned short*)alloc((size_t)BATCH * NHEAD * NTOK * DHEAD * 2);
  unsigned short* Vt  = (unsigned short*)alloc((size_t)BATCH * NHEAD * NTOK * DHEAD * 2);
  unsigned short* ffb = (unsigned short*)alloc((size_t)MROWS * FFDIM * 2);
  unsigned short* WqkvB = (unsigned short*)alloc((size_t)NLAYER * 3 * DMODEL * DMODEL * 2);
  unsigned short* WoB   = (unsigned short*)alloc((size_t)NLAYER * DMODEL * DMODEL * 2);
  unsigned short* W1B   = (unsigned short*)alloc((size_t)NLAYER * FFDIM * DMODEL * 2);
  unsigned short* W2B   = (unsigned short*)alloc((size_t)NLAYER * DMODEL * FFDIM * 2);

  auto cvt = [&](const float* s, unsigned short* d, int n) {
    cvt_kernel<<<(n + 255) / 256, 256, 0, stream>>>(s, d, n);
  };
  cvt(Wqkv, WqkvB, NLAYER * 3 * DMODEL * DMODEL);
  cvt(Wo, WoB, NLAYER * DMODEL * DMODEL);
  cvt(W1, W1B, NLAYER * FFDIM * DMODEL);
  cvt(W2, W2B, NLAYER * DMODEL * FFDIM);

  embed_kernel<<<MROWS * DMODEL / 256, 256, 0, stream>>>(x, cls1, cls2, ind, h, hb);

  for (int l = 0; l < NLAYER; ++l) {
    // QKV projection (scatter into Q, K, V^T bf16)
    gemm_kernel<2><<<dim3(12, 32), 256, 0, stream>>>(
        hb, WqkvB + (size_t)l * 3 * DMODEL * DMODEL, bqkv + (size_t)l * 3 * DMODEL,
        DMODEL, 3 * DMODEL, nullptr, nullptr, Qb, Kb, Vt);
    // flash attention
    attention_kernel<<<dim3(NTOK / 128, NHEAD, BATCH), 256, 0, stream>>>(
        Qb, Kb, Vt, ind, ab);
    // output projection
    gemm_kernel<0><<<dim3(4, 32), 256, 0, stream>>>(
        ab, WoB + (size_t)l * DMODEL * DMODEL, bo + (size_t)l * DMODEL,
        DMODEL, DMODEL, ybuf, nullptr, nullptr, nullptr, nullptr);
    add_ln_kernel<<<MROWS, 256, 0, stream>>>(h, ybuf, ln1g + (size_t)l * DMODEL,
                                             ln1b + (size_t)l * DMODEL, hb);
    // FFN
    gemm_kernel<1><<<dim3(32, 32), 256, 0, stream>>>(
        hb, W1B + (size_t)l * FFDIM * DMODEL, b1 + (size_t)l * FFDIM,
        DMODEL, FFDIM, nullptr, ffb, nullptr, nullptr, nullptr);
    gemm_kernel<0><<<dim3(4, 32), 256, 0, stream>>>(
        ffb, W2B + (size_t)l * DMODEL * FFDIM, b2 + (size_t)l * DMODEL,
        FFDIM, DMODEL, ybuf, nullptr, nullptr, nullptr, nullptr);
    add_ln_kernel<<<MROWS, 256, 0, stream>>>(h, ybuf, ln2g + (size_t)l * DMODEL,
                                             ln2b + (size_t)l * DMODEL, hb);
  }
  gather_kernel<<<8, 256, 0, stream>>>(h, ind, (float*)d_out);
}
